// QuantizedLinear_80994493268426
// MI455X (gfx1250) — matrix operand order repaired and checked by reading, not compiled, not run on hardware
//
#include <hip/hip_runtime.h>

// ---------------------------------------------------------------------------
// QuantizedLinear (TWN-style ternary weight) for gfx1250 / MI455X.
//   out = scale * (x @ T^T) + bias,  T in {-1,0,+1} (exact in bf16)
//   x is split into hi/lo bf16 terms -> fp32-accurate via two WMMAs per tile.
// ---------------------------------------------------------------------------

typedef __attribute__((ext_vector_type(16))) __bf16 v16bf;
typedef __attribute__((ext_vector_type(8)))  __bf16 v8bf;
typedef __attribute__((ext_vector_type(8)))  float  v8f;

#define D_IN   512
#define D_OUT  512
#define EPS_Q  1e-8f
#define THRESH 0.7f

// ---------------------------- Stage 1a: partial |w| sums --------------------
__global__ __launch_bounds__(256)
void qk_abs_partial(const float* __restrict__ w, float* __restrict__ partials) {
    __shared__ float sdata[256];
    const int tid = threadIdx.x;
    const int base = blockIdx.x * 1024 + tid;   // 256 blocks * 1024 elems
    float s = 0.f;
#pragma unroll
    for (int i = 0; i < 4; ++i) s += __builtin_fabsf(w[base + i * 256]);
    sdata[tid] = s;
    __syncthreads();
    for (int off = 128; off > 0; off >>= 1) {
        if (tid < off) sdata[tid] += sdata[tid + off];
        __syncthreads();
    }
    if (tid == 0) partials[blockIdx.x] = sdata[0];
}

// ---------------------------- Stage 1b: finalize scale ----------------------
__global__ __launch_bounds__(256)
void qk_finalize(const float* __restrict__ partials, float* __restrict__ scalars) {
    __shared__ float sdata[256];
    const int tid = threadIdx.x;
    sdata[tid] = partials[tid];
    __syncthreads();
    for (int off = 128; off > 0; off >>= 1) {
        if (tid < off) sdata[tid] += sdata[tid + off];
        __syncthreads();
    }
    if (tid == 0) {
        float scale = sdata[0] * (1.0f / (float)(D_IN * D_OUT));
        scale = fmaxf(scale, EPS_Q);
        scalars[0] = scale;
        scalars[1] = THRESH * scale;
    }
}

// ---------------------------- Stage 2: ternarize to bf16 --------------------
// Tq keeps weight's [D_OUT][D_IN] row-major layout: row = output feature (N),
// contiguous in K. That is exactly the per-lane "one column of B" order the
// WMMA B-fragment wants.
__global__ __launch_bounds__(256)
void qk_ternarize(const float* __restrict__ w, const float* __restrict__ scalars,
                  __bf16* __restrict__ Tq) {
    const float thr = scalars[1];
    const int i = blockIdx.x * 256 + threadIdx.x;
    const float v = w[i];
    float t = 0.f;
    if (__builtin_fabsf(v) > thr) t = (v > 0.f) ? 1.0f : -1.0f;
    Tq[i] = (__bf16)t;   // -1, 0, +1 are exact in bf16
}

// ---------------------------- Stage 3: WMMA GEMM ----------------------------
// Block: 512 threads = 16 waves, arranged 4 (M) x 4 (N).
// Block tile: 64 M x 512 N  (full N per block -> x streamed from HBM once).
// Wave tile : 16 M x 128 N  -> 8 accumulators (v8f).
__global__ __launch_bounds__(512)
void qk_gemm(const float* __restrict__ x, const __bf16* __restrict__ Tq,
             const float* __restrict__ bias, const float* __restrict__ scalars,
             float* __restrict__ out) {
    const int lane     = threadIdx.x & 31;
    const int wave     = threadIdx.x >> 5;
    const int wm       = wave >> 2;          // 0..3 (M direction)
    const int wn       = wave & 3;           // 0..3 (N direction)
    const int lane15   = lane & 15;
    const int laneHalf = lane >> 4;          // 0 or 1

    const int mBase = blockIdx.x * 64 + wm * 16;
    const int nWave = wn * 128;

    const float scale = scalars[0];

    // A-fragment row owned by this lane (ISA: lanes 0-15 and 16-31 both map M=0..15)
    const float* xRow = x + (size_t)(mBase + lane15) * D_IN;

    v8f acc[8];
#pragma unroll
    for (int i = 0; i < 8; ++i) acc[i] = {};

    for (int k0 = 0; k0 < D_IN; k0 += 32) {
        // -------- A fragment: 16 f32 per lane, contiguous K order, same as B ----
        // lane<16  : K = k0 + {0..15}
        // lane>=16 : K = k0 + {16..31}
        const int kOff = laneHalf * 16;
        const float4 a0 = *(const float4*)(xRow + k0 + kOff + 0);
        const float4 a1 = *(const float4*)(xRow + k0 + kOff + 4);
        const float4 a2 = *(const float4*)(xRow + k0 + kOff + 8);
        const float4 a3 = *(const float4*)(xRow + k0 + kOff + 12);
        float af[16] = {a0.x, a0.y, a0.z, a0.w, a1.x, a1.y, a1.z, a1.w,
                        a2.x, a2.y, a2.z, a2.w, a3.x, a3.y, a3.z, a3.w};

        v16bf ahi, alo;
#pragma unroll
        for (int j = 0; j < 16; ++j) {
            const __bf16 h = (__bf16)af[j];
            ahi[j] = h;
            alo[j] = (__bf16)(af[j] - (float)h);   // residual term
        }

        // prefetch next k-slab of x
        __builtin_prefetch(xRow + k0 + 32, 0, 0);

        // -------- 8 B fragments along N; 2 WMMAs each (hi + lo) --------------
#pragma unroll
        for (int nt = 0; nt < 8; ++nt) {
            const int nCol = nWave + nt * 16 + lane15;
            // lane = one row of T (column of B); 16 contiguous bf16 along K
            const __bf16* bp = Tq + (size_t)nCol * D_IN + k0 + laneHalf * 16;
            const v8bf b0 = *(const v8bf*)(bp);
            const v8bf b1 = *(const v8bf*)(bp + 8);
            const v16bf bfrag = __builtin_shufflevector(
                b0, b1, 0, 1, 2, 3, 4, 5, 6, 7, 8, 9, 10, 11, 12, 13, 14, 15);

            acc[nt] = __builtin_amdgcn_wmma_f32_16x16x32_bf16(
                false, ahi, false, bfrag, (short)0, acc[nt], false, false);
            acc[nt] = __builtin_amdgcn_wmma_f32_16x16x32_bf16(
                false, alo, false, bfrag, (short)0, acc[nt], false, false);
        }
    }

    // -------- epilogue: scale + bias, per C/D layout --------------------------
    // acc[nt][r] holds (M = mBase + r + 8*laneHalf, N = nWave + nt*16 + lane15)
    const int mRow = mBase + laneHalf * 8;
#pragma unroll
    for (int nt = 0; nt < 8; ++nt) {
        const int nCol = nWave + nt * 16 + lane15;
        const float b = bias[nCol];
        float* op = out + (size_t)mRow * D_OUT + nCol;
#pragma unroll
        for (int r = 0; r < 8; ++r) {
            op[(size_t)r * D_OUT] = scale * acc[nt][r] + b;
        }
    }
}

// ---------------------------------------------------------------------------
extern "C" void kernel_launch(void* const* d_in, const int* in_sizes, int n_in,
                              void* d_out, int out_size, void* d_ws, size_t ws_size,
                              hipStream_t stream) {
    const float* x      = (const float*)d_in[0];   // [B,S,D_IN] f32
    const float* weight = (const float*)d_in[1];   // [D_OUT,D_IN] f32
    const float* bias   = (const float*)d_in[2];   // [D_OUT] f32
    float*       out    = (float*)d_out;           // [B,S,D_OUT] f32

    // workspace layout: [Tq bf16 512KB][partials 1KB][scalars]
    char*   ws       = (char*)d_ws;
    __bf16* Tq       = (__bf16*)ws;
    float*  partials = (float*)(ws + (size_t)D_OUT * D_IN * sizeof(__bf16));
    float*  scalars  = (float*)(ws + (size_t)D_OUT * D_IN * sizeof(__bf16) + 1024);

    const int M = in_sizes[0] / D_IN;              // 65536 rows of x

    qk_abs_partial<<<256, 256, 0, stream>>>(weight, partials);
    qk_finalize  <<<1,   256, 0, stream>>>(partials, scalars);
    qk_ternarize <<<(D_OUT * D_IN) / 256, 256, 0, stream>>>(weight, scalars, Tq);
    qk_gemm      <<<M / 64, 512, 0, stream>>>(x, Tq, bias, scalars, out);
}
